// GraphS4mer_Regression_80023830659663
// MI455X (gfx1250) — matrix operand order repaired and checked by reading, not compiled, not run on hardware
//
#include <hip/hip_runtime.h>

typedef __attribute__((ext_vector_type(16))) __bf16 v16bf;
typedef __attribute__((ext_vector_type(8)))  float  v8f;

constexpr int kB = 16, kN = 19, kL = 1000, kRES = 100, kDIN = 4;
constexpr int kH = 128, kG3 = 384, kKE = 180, kBN = 304;
constexpr int kWP = 132;          // bf16 LDS row pitch (bank-spread padding)
constexpr float kNEG = 0.01f;     // LeakyReLU slope

__device__ __forceinline__ unsigned short f2bf(float f) {
  unsigned u = __float_as_uint(f);
  u += 0x7FFFu + ((u >> 16) & 1u);          // round-to-nearest-even
  return (unsigned short)(u >> 16);
}

// A-matrix 16x32 bf16 fragment: lanes 0-15 K=0..15, lanes 16-31 K=16..31,
// VGPR v holds K pair (v*2).
__device__ __forceinline__ v16bf fragA(const unsigned short* row, int kstart) {
  union { v16bf vec; unsigned u[8]; } f;
#pragma unroll
  for (int i = 0; i < 8; ++i) {
    int k = i * 2;
    f.u[i] = *(const unsigned*)(row + kstart + k);
  }
  return f.vec;
}
// B-matrix 32x16 bf16 fragment: lanes 0-15 K=0..15, lanes 16-31 K=16..31,
// VGPR v holds K pair (v*2). `row` points at B^T row n (= column n of B).
__device__ __forceinline__ v16bf fragB(const unsigned short* row, int kstart) {
  union { v16bf vec; unsigned u[8]; } f;
#pragma unroll
  for (int i = 0; i < 8; ++i)
    f.u[i] = *(const unsigned*)(row + kstart + i * 2);
  return f.vec;
}

// ---------------------------------------------------------------------------
// Kernel 1: GRU over time. 19 WGs x 16 sequences, Whh resident in LDS (bf16),
// B fragments resident in VGPRs across all 1000 steps. 12 WMMAs / wave / step.
// Emits h at t=L-1 and mean(h) over the last RES window.
// ---------------------------------------------------------------------------
extern "C" __global__ void __launch_bounds__(256)
gru_fwd(const float* __restrict__ x, const float* __restrict__ Wih,
        const float* __restrict__ Whh, const float* __restrict__ bih,
        const float* __restrict__ bhh, float* __restrict__ hlast,
        float* __restrict__ hmean) {
  constexpr int SEQ = 16;
  extern __shared__ __align__(16) char smem[];
  unsigned short* sWhh = (unsigned short*)smem;        // [kG3][kWP] bf16
  unsigned short* sHbf = sWhh + kG3 * kWP;             // [SEQ][kWP] bf16
  float* sH    = (float*)(sHbf + SEQ * kWP);           // [SEQ][kH]
  float* sHsum = sH + SEQ * kH;                        // [SEQ][kH]
  float* sGate = sHsum + SEQ * kH;                     // [SEQ][kG3]
  float* sWih  = sGate + SEQ * kG3;                    // [kG3][kDIN]
  float* sBih  = sWih + kG3 * kDIN;                    // [kG3]
  float* sBhh  = sBih + kG3;                           // [kG3]
  float* sX    = sBhh + kG3;                           // [SEQ][kDIN]

  const int tid  = threadIdx.x;
  const int seq0 = blockIdx.x * SEQ;

  for (int i = tid; i < kG3 * kH; i += 256) {
    int g = i >> 7, k = i & (kH - 1);
    sWhh[g * kWP + k] = f2bf(Whh[i]);
  }
  for (int i = tid; i < kG3 * (kWP - kH); i += 256) {
    int g = i / (kWP - kH), k = kH + i % (kWP - kH);
    sWhh[g * kWP + k] = 0;
  }
  for (int i = tid; i < kG3 * kDIN; i += 256) sWih[i] = Wih[i];
  for (int i = tid; i < kG3; i += 256) { sBih[i] = bih[i]; sBhh[i] = bhh[i]; }
  for (int i = tid; i < SEQ * kH; i += 256) { sH[i] = 0.f; sHsum[i] = 0.f; }
  for (int i = tid; i < SEQ * kWP; i += 256) sHbf[i] = 0;
  __syncthreads();

  const int wave = tid >> 5, lane = tid & 31;
  const int lrow = lane & 15, lhi = lane >> 4;

  // Preload B fragments (Whh^T): wave w owns n-tiles {w, w+8, w+16}; constant
  // over time -> 96 VGPRs resident for the whole recurrence.
  v16bf Bf[3][4];
#pragma unroll
  for (int j = 0; j < 3; ++j) {
    const unsigned short* row = sWhh + ((wave + j * 8) * 16 + lrow) * kWP;
#pragma unroll
    for (int ks = 0; ks < 4; ++ks)
      Bf[j][ks] = fragB(row, ks * 32 + lhi * 16);
  }

  for (int t = 0; t < kL; ++t) {
    if (tid < SEQ * kDIN) {
      int m = tid >> 2, d = tid & 3;
      sX[tid] = x[(size_t)(seq0 + m) * (kL * kDIN) + t * kDIN + d];
    }
    for (int i = tid; i < SEQ * kH; i += 256) {
      int m = i >> 7, k = i & (kH - 1);
      sHbf[m * kWP + k] = f2bf(sH[i]);
    }
    __syncthreads();

    v16bf Af[4];
    {
      const unsigned short* row = sHbf + lrow * kWP;
#pragma unroll
      for (int ks = 0; ks < 4; ++ks)
        Af[ks] = fragA(row, ks * 32 + lhi * 16);
    }
#pragma unroll
    for (int j = 0; j < 3; ++j) {
      v8f acc = {0.f, 0.f, 0.f, 0.f, 0.f, 0.f, 0.f, 0.f};
#pragma unroll
      for (int ks = 0; ks < 4; ++ks)
        acc = __builtin_amdgcn_wmma_f32_16x16x32_bf16(
            false, Af[ks], false, Bf[j][ks], (short)0, acc, false, false);
      int g0 = (wave + j * 8) * 16 + lrow;
#pragma unroll
      for (int vv = 0; vv < 8; ++vv)
        sGate[(vv + lhi * 8) * kG3 + g0] = acc[vv];
    }
    __syncthreads();

    const bool accum = (t >= kL - kRES);
#pragma unroll
    for (int e = 0; e < (SEQ * kH) / 256; ++e) {
      int idx = tid + e * 256;
      int m = idx >> 7, j = idx & (kH - 1);
      float xr = sBih[j], xz = sBih[j + kH], xn = sBih[j + 2 * kH];
#pragma unroll
      for (int d = 0; d < kDIN; ++d) {
        float xv = sX[m * kDIN + d];
        xr += xv * sWih[j * kDIN + d];
        xz += xv * sWih[(j + kH) * kDIN + d];
        xn += xv * sWih[(j + 2 * kH) * kDIN + d];
      }
      float hr = sGate[m * kG3 + j] + sBhh[j];
      float hz = sGate[m * kG3 + j + kH] + sBhh[j + kH];
      float hn = sGate[m * kG3 + j + 2 * kH] + sBhh[j + 2 * kH];
      float r = 1.f / (1.f + __expf(-(xr + hr)));
      float z = 1.f / (1.f + __expf(-(xz + hz)));
      float n = tanhf(xn + r * hn);
      float h = (1.f - z) * n + z * sH[idx];
      sH[idx] = h;
      if (accum) sHsum[idx] += h;
    }
    __syncthreads();
  }

  for (int i = tid; i < SEQ * kH; i += 256) {
    hlast[(size_t)seq0 * kH + i] = sH[i];
    hmean[(size_t)seq0 * kH + i] = sHsum[i] * (1.f / kRES);
  }
}

// ---------------------------------------------------------------------------
// Kernel 2: graph learner for the LAST window only (the only one the output
// depends on). One WG per batch. q/k GEMMs via WMMA (19 rows padded to 32).
// ---------------------------------------------------------------------------
extern "C" __global__ void __launch_bounds__(256)
graph_learn(const float* __restrict__ hmean, const float* __restrict__ Wq,
            const float* __restrict__ Wk, float* __restrict__ maskw,
            float* __restrict__ degw) {
  extern __shared__ __align__(16) char smem[];
  float* sXf   = (float*)smem;               // [kN][kH]
  float* sXn   = sXf + kN * kH;
  float* sQ    = sXn + kN * kH;
  float* sK    = sQ + kN * kH;
  float* sDist = sK + kN * kH;               // [361]
  float* sAdjK = sDist + kN * kN;
  float* sAttn = sAdjK + kN * kN;
  float* sAdj  = sAttn + kN * kN;
  float* sCand = sAdj + kN * kN;             // [512]
  unsigned short* sXbf = (unsigned short*)(sCand + 512);  // [32][kWP]
  unsigned short* sWqT = sXbf + 32 * kWP;                 // [kH][kWP]
  unsigned short* sWkT = sWqT + kH * kWP;

  const int tid = threadIdx.x, b = blockIdx.x;
  const int wave = tid >> 5, lane = tid & 31;
  const int lrow = lane & 15, lhi = lane >> 4;

  for (int i = tid; i < kN * kH; i += 256)
    sXf[i] = hmean[(size_t)(b * kN) * kH + i];
  for (int i = tid; i < 32 * kWP; i += 256) sXbf[i] = 0;
  for (int i = tid; i < kN * kN; i += 256) sAdjK[i] = 0.f;
  __syncthreads();

  for (int i = tid; i < kN * kH; i += 256) {
    int m = i >> 7, k = i & (kH - 1);
    sXbf[m * kWP + k] = f2bf(sXf[i]);
  }
  for (int i = tid; i < kH * kH; i += 256) {
    int k = i >> 7, n = i & (kH - 1);
    sWqT[n * kWP + k] = f2bf(Wq[i]);
    sWkT[n * kWP + k] = f2bf(Wk[i]);
  }
  if (tid < kN) {
    float ss = 0.f;
    for (int d = 0; d < kH; ++d) { float v = sXf[tid * kH + d]; ss += v * v; }
    float inv = 1.f / (sqrtf(ss) + 1e-12f);
    for (int d = 0; d < kH; ++d) sXn[tid * kH + d] = sXf[tid * kH + d] * inv;
  }
  __syncthreads();

  for (int e = tid; e < kN * kN; e += 256) {
    int n = e / kN, m = e % kN;
    float s = 0.f;
    for (int d = 0; d < kH; ++d) s += sXn[n * kH + d] * sXn[m * kH + d];
    sDist[e] = s;
  }
  __syncthreads();

  if (tid < kN) {  // top-3 per row, scatter similarity values
    float v0 = -1e30f, v1 = -1e30f, v2 = -1e30f;
    int i0 = 0, i1 = 0, i2 = 0;
    for (int m = 0; m < kN; ++m) {
      float v = sDist[tid * kN + m];
      if (v > v0) { v2 = v1; i2 = i1; v1 = v0; i1 = i0; v0 = v; i0 = m; }
      else if (v > v1) { v2 = v1; i2 = i1; v1 = v; i1 = m; }
      else if (v > v2) { v2 = v; i2 = m; }
    }
    sAdjK[tid * kN + i0] = v0;
    sAdjK[tid * kN + i1] = v1;
    sAdjK[tid * kN + i2] = v2;
  }
  __syncthreads();

  // q = x_ @ Wq, k = x_ @ Wk  (wave w -> n-tile w; 2 M-tiles; 4 k-steps)
  {
    const unsigned short* bRowQ = sWqT + (wave * 16 + lrow) * kWP;
    const unsigned short* bRowK = sWkT + (wave * 16 + lrow) * kWP;
#pragma unroll
    for (int mt = 0; mt < 2; ++mt) {
      const unsigned short* aRow = sXbf + (mt * 16 + lrow) * kWP;
      v8f aq = {0.f, 0.f, 0.f, 0.f, 0.f, 0.f, 0.f, 0.f};
      v8f ak = {0.f, 0.f, 0.f, 0.f, 0.f, 0.f, 0.f, 0.f};
#pragma unroll
      for (int ks = 0; ks < 4; ++ks) {
        v16bf A = fragA(aRow, ks * 32 + lhi * 16);
        aq = __builtin_amdgcn_wmma_f32_16x16x32_bf16(
            false, A, false, fragB(bRowQ, ks * 32 + lhi * 16), (short)0, aq, false, false);
        ak = __builtin_amdgcn_wmma_f32_16x16x32_bf16(
            false, A, false, fragB(bRowK, ks * 32 + lhi * 16), (short)0, ak, false, false);
      }
      int n = wave * 16 + lrow;
#pragma unroll
      for (int vv = 0; vv < 8; ++vv) {
        int m = mt * 16 + vv + lhi * 8;
        if (m < kN) { sQ[m * kH + n] = aq[vv]; sK[m * kH + n] = ak[vv]; }
      }
    }
  }
  __syncthreads();

  if (tid < kN) {  // attention row softmax
    float srow[kN];
    float mx = -1e30f;
    for (int m = 0; m < kN; ++m) {
      float s = 0.f;
      for (int d = 0; d < kH; ++d) s += sQ[tid * kH + d] * sK[m * kH + d];
      s *= 0.08838834764831845f;  // 1/sqrt(128)
      srow[m] = s; mx = fmaxf(mx, s);
    }
    float sum = 0.f;
    for (int m = 0; m < kN; ++m) { srow[m] = __expf(srow[m] - mx); sum += srow[m]; }
    float inv = 1.f / sum;
    for (int m = 0; m < kN; ++m) sAttn[tid * kN + m] = srow[m] * inv;
  }
  __syncthreads();

  for (int e = tid; e < kN * kN; e += 256) {
    int n = e / kN, m = e % kN;
    float ak = 0.5f * (fmaxf(sAdjK[n * kN + m], 0.f) + fmaxf(sAdjK[m * kN + n], 0.f));
    if (n == m) ak = 1.f;
    float at = 0.5f * (sAttn[n * kN + m] + sAttn[m * kN + n]);
    sAdj[e] = 0.2f * ak + 0.8f * at;
  }
  __syncthreads();

  // thr = (KE+1)-th largest = min{ v : #(elements > v) <= KE }
  for (int i = tid; i < 512; i += 256) sCand[i] = 3.4e38f;
  __syncthreads();
  for (int e = tid; e < kN * kN; e += 256) {
    float v = sAdj[e];
    int gt = 0;
    for (int i = 0; i < kN * kN; ++i) gt += (sAdj[i] > v) ? 1 : 0;
    if (gt <= kKE) sCand[e] = v;
  }
  __syncthreads();
  for (int s2 = 256; s2 >= 1; s2 >>= 1) {
    if (tid < s2) sCand[tid] = fminf(sCand[tid], sCand[tid + s2]);
    __syncthreads();
  }
  float thr = sCand[0];

  for (int e = tid; e < kN * kN; e += 256) {
    int n = e / kN, m = e % kN;
    float a = (sAdj[e] > thr) ? sAdj[e] : 0.f;
    float al = (n == m) ? 1.f : a;        // self loops weight 1
    sAdj[e] = (al != 0.f) ? 1.f : 0.f;    // binary mask
  }
  __syncthreads();
  for (int e = tid; e < kN * kN; e += 256)
    maskw[(size_t)b * kN * kN + e] = sAdj[e];
  if (tid < kN) {
    float dg = 0.f;
    for (int m = 0; m < kN; ++m) dg += sAdj[tid * kN + m];
    degw[b * kN + tid] = dg;
  }
}

// ---------------------------------------------------------------------------
// Kernel 3: 2x GraphSAGE (mean) + decoder at t = L-1 only. One WG per batch.
// aggr@Wl and xg@Wr fused into one WMMA accumulator chain per tile.
// ---------------------------------------------------------------------------
extern "C" __global__ void __launch_bounds__(256)
sage_decode(const float* __restrict__ hlast, const float* __restrict__ maskw,
            const float* __restrict__ degw, const float* __restrict__ Wl,
            const float* __restrict__ bl, const float* __restrict__ Wr,
            const float* __restrict__ Wdec, const float* __restrict__ bdec,
            float* __restrict__ out) {
  extern __shared__ __align__(16) char smem[];
  float* sXg   = (float*)smem;              // [kN][kH]
  float* sAg   = sXg + kN * kH;             // [kN][kH]
  float* sMask = sAg + kN * kH;             // [361]
  float* sDeg  = sMask + kN * kN;           // [20]
  unsigned short* sXbf = (unsigned short*)(sDeg + 20);  // [32][kWP]
  unsigned short* sAbf = sXbf + 32 * kWP;               // [32][kWP]
  unsigned short* sWlT = sAbf + 32 * kWP;               // [kH][kWP]
  unsigned short* sWrT = sWlT + kH * kWP;               // [kH][kWP]

  const int tid = threadIdx.x, b = blockIdx.x;
  const int wave = tid >> 5, lane = tid & 31;
  const int lrow = lane & 15, lhi = lane >> 4;

  for (int i = tid; i < kN * kH; i += 256)
    sXg[i] = hlast[(size_t)(b * kN) * kH + i];
  for (int i = tid; i < kN * kN; i += 256)
    sMask[i] = maskw[(size_t)b * kN * kN + i];
  if (tid < kN) sDeg[tid] = degw[b * kN + tid];
  for (int i = tid; i < 32 * kWP; i += 256) { sXbf[i] = 0; sAbf[i] = 0; }
  __syncthreads();

  for (int layer = 0; layer < 2; ++layer) {
    for (int i = tid; i < kN * kH; i += 256) {  // mean aggregation
      int n = i >> 7, d = i & (kH - 1);
      float s = 0.f;
      for (int m = 0; m < kN; ++m) s += sMask[n * kN + m] * sXg[m * kH + d];
      sAg[i] = s / sDeg[n];
    }
    __syncthreads();
    for (int i = tid; i < kN * kH; i += 256) {
      int m = i >> 7, k = i & (kH - 1);
      sXbf[m * kWP + k] = f2bf(sXg[i]);
      sAbf[m * kWP + k] = f2bf(sAg[i]);
    }
    for (int i = tid; i < kH * kH; i += 256) {
      int k = i >> 7, n = i & (kH - 1);
      sWlT[n * kWP + k] = f2bf(Wl[layer * kH * kH + i]);
      sWrT[n * kWP + k] = f2bf(Wr[layer * kH * kH + i]);
    }
    __syncthreads();

    v8f acc[2];
    const unsigned short* bRowL = sWlT + (wave * 16 + lrow) * kWP;
    const unsigned short* bRowR = sWrT + (wave * 16 + lrow) * kWP;
#pragma unroll
    for (int mt = 0; mt < 2; ++mt) {
      const unsigned short* aRowX = sXbf + (mt * 16 + lrow) * kWP;
      const unsigned short* aRowA = sAbf + (mt * 16 + lrow) * kWP;
      v8f a = {0.f, 0.f, 0.f, 0.f, 0.f, 0.f, 0.f, 0.f};
#pragma unroll
      for (int ks = 0; ks < 4; ++ks) {
        a = __builtin_amdgcn_wmma_f32_16x16x32_bf16(
            false, fragA(aRowA, ks * 32 + lhi * 16), false,
            fragB(bRowL, ks * 32 + lhi * 16), (short)0, a, false, false);
        a = __builtin_amdgcn_wmma_f32_16x16x32_bf16(
            false, fragA(aRowX, ks * 32 + lhi * 16), false,
            fragB(bRowR, ks * 32 + lhi * 16), (short)0, a, false, false);
      }
      acc[mt] = a;
    }
    __syncthreads();
    int n = wave * 16 + lrow;
    float bias = bl[layer * kH + n];
#pragma unroll
    for (int mt = 0; mt < 2; ++mt) {
#pragma unroll
      for (int vv = 0; vv < 8; ++vv) {
        int m = mt * 16 + vv + lhi * 8;
        if (m < kN) {
          float v = acc[mt][vv] + bias;
          sXg[m * kH + n] = (v > 0.f) ? v : kNEG * v;
        }
      }
    }
    __syncthreads();
  }

  if (tid < kN) {  // decoder: last timestep, l_output=1
    float s = 0.f;
    for (int d = 0; d < kH; ++d) s += sXg[tid * kH + d] * Wdec[d];
    out[b * kN + tid] = s + bdec[0];
  }
}

// ---------------------------------------------------------------------------
extern "C" void kernel_launch(void* const* d_in, const int* in_sizes, int n_in,
                              void* d_out, int out_size, void* d_ws, size_t ws_size,
                              hipStream_t stream) {
  (void)in_sizes; (void)n_in; (void)out_size; (void)ws_size;
  const float* x    = (const float*)d_in[0];
  const float* Wih  = (const float*)d_in[1];
  const float* Whh  = (const float*)d_in[2];
  const float* bih  = (const float*)d_in[3];
  const float* bhh  = (const float*)d_in[4];
  const float* Wq   = (const float*)d_in[5];
  const float* Wk   = (const float*)d_in[6];
  const float* sWl  = (const float*)d_in[7];
  const float* sbl  = (const float*)d_in[8];
  const float* sWr  = (const float*)d_in[9];
  const float* Wdec = (const float*)d_in[10];
  const float* bdec = (const float*)d_in[11];
  float* out = (float*)d_out;

  float* ws    = (float*)d_ws;
  float* hlast = ws;                        // [304][128]
  float* hmean = hlast + kBN * kH;          // [304][128]
  float* maskw = hmean + kBN * kH;          // [16][19][19]
  float* degw  = maskw + kB * kN * kN;      // [16][19]

  size_t smem1 = (size_t)(kG3 * kWP + 16 * kWP) * 2 +
                 (size_t)(16 * kH * 2 + 16 * kG3 + kG3 * kDIN + 2 * kG3 + 16 * kDIN) * 4;
  size_t smem2 = (size_t)(4 * kN * kH + 4 * kN * kN + 512) * 4 +
                 (size_t)(32 * kWP + 2 * kH * kWP) * 2;
  size_t smem3 = (size_t)(2 * kN * kH + kN * kN + 20) * 4 +
                 (size_t)(2 * 32 * kWP + 2 * kH * kWP) * 2;

  gru_fwd<<<kBN / 16, 256, smem1, stream>>>(x, Wih, Whh, bih, bhh, hlast, hmean);
  graph_learn<<<kB, 256, smem2, stream>>>(hmean, Wq, Wk, maskw, degw);
  sage_decode<<<kB, 256, smem3, stream>>>(hlast, maskw, degw, sWl, sbl, sWr,
                                          Wdec, bdec, out);
}
